// GGNN_49649821941998
// MI455X (gfx1250) — matrix operand order repaired and checked by reading, not compiled, not run on hardware
//
#include <hip/hip_runtime.h>
#include <hip/hip_bf16.h>
#include <cstdint>
#include <cstddef>

typedef __attribute__((ext_vector_type(16))) _Float16 v16h;
typedef __attribute__((ext_vector_type(8)))  float    v8f;

#define NN 1024
#define DD 64
#define IN_DIM 32
#define EE 32768

__device__ __forceinline__ float sigm(float x) { return 1.0f / (1.0f + __expf(-x)); }
__device__ __forceinline__ float leaky(float x) { return x > 0.f ? x : 0.01f * x; }

// ---------------------------------------------------------------------------
// WMMA A-fragment packing (CDNA5 ISA 7.12.2, 16-bit A 16x32):
//   lane L: M = L%16 ; elem e: K = (e<8) ? 8*(L/16)+e : 8+8*(L/16)+e
// Packed linear layout: [((mtile*ksteps + ks)*32 + lane)*16 + e]
// Inverse map (n,d) -> packed index, K = row length (32 or 64):
// ---------------------------------------------------------------------------
__device__ __forceinline__ size_t a_pack_idx(int n, int d, int K) {
  int mtile = n >> 4, lr = n & 15;
  int ks = d >> 5, kk = d & 31;
  int half, e;
  if (kk < 16) { half = kk >> 3;        e = kk & 7; }
  else         { half = (kk - 16) >> 3; e = ((kk - 16) & 7) + 8; }
  int lane = half * 16 + lr;
  return ((size_t)(mtile * (K >> 5) + ks) * 32 + lane) * 16 + e;
}

// ---------------------------------------------------------------------------
// Pack B[K,N] f32 -> f16 fragment order (B 32x16: lane L -> N=L%16,
// elem e -> K = (e<8) ? 8*(L/16)+e : 16+8*(L/16)+e-8). transposed=1 means src is [N,K] (w_ih / w_hh).
// dst index i = ((ntile*ksteps + kstep)*32 + lane)*16 + e
// ---------------------------------------------------------------------------
__global__ void pack_B(_Float16* __restrict__ dst, const float* __restrict__ src,
                       int K, int N, int transposed) {
  int i = blockIdx.x * blockDim.x + threadIdx.x;
  if (i >= K * N) return;
  int e     = i & 15;
  int lane  = (i >> 4) & 31;
  int rest  = i >> 9;
  int ksteps = K >> 5;
  int kstep = rest % ksteps;
  int ntile = rest / ksteps;
  int k = kstep * 32 + ((e < 8) ? 8 * (lane >> 4) + e : 16 + 8 * (lane >> 4) + (e - 8));
  int n = ntile * 16 + (lane & 15);
  float v = transposed ? src[(size_t)n * K + k] : src[(size_t)k * N + n];
  dst[i] = (_Float16)v;
}

// ---------------------------------------------------------------------------
// WMMA GEMM: C[M,N] = act(A @ B + bias). A and B pre-packed in fragment
// order -> inner loop is 4x global_load_b128 + 1x v_wmma per K-step.
// 4 waves per block, one 16x16 tile per wave.
// C/D layout: lane L -> N=L%16 ; vgpr r -> M = r + 8*(L/16).
// Optionally mirrors output as f16 in A-fragment order (K=64 consumers).
// ---------------------------------------------------------------------------
__global__ void gemm_wmma_f16(const _Float16* __restrict__ Ap,
                              const _Float16* __restrict__ Bp,
                              const float* __restrict__ bias,
                              float* __restrict__ C,
                              _Float16* __restrict__ C16p,
                              int M, int N, int K, int act) {
  const int lane = threadIdx.x & 31;
  const int wid  = threadIdx.x >> 5;
  const int tile = blockIdx.x * 4 + wid;
  const int ntn  = N >> 4;
  if (tile >= (M >> 4) * ntn) return;       // uniform per wave
  const int tmt = tile / ntn;               // m-tile index
  const int tnt = tile % ntn;               // n-tile index
  const int ksteps = K >> 5;

  const v16h* __restrict__ ap =
      (const v16h*)(Ap + ((size_t)tmt * ksteps * 32 + lane) * 16);
  const v16h* __restrict__ bp =
      (const v16h*)(Bp + ((size_t)tnt * ksteps * 32 + lane) * 16);

  v8f acc = {};
  for (int ks = 0; ks < ksteps; ++ks) {
    v16h a = ap[(size_t)ks * 32];
    v16h b = bp[(size_t)ks * 32];
    acc = __builtin_amdgcn_wmma_f32_16x16x32_f16(false, a, false, b,
                                                 (short)0, acc, false, false);
  }

  const int halfid = lane >> 4;
  const int lr     = lane & 15;
  const int col    = (tnt << 4) + lr;
  const float bv   = bias ? bias[col] : 0.0f;
#pragma unroll
  for (int r = 0; r < 8; ++r) {
    const int row = (tmt << 4) + r + 8 * halfid;
    float v = acc[r] + bv;
    if (act == 1) v = leaky(v);
    C[(size_t)row * N + col] = v;
    if (C16p) C16p[a_pack_idx(row, col, DD)] = (_Float16)v;
  }
}

// ---------------------------------------------------------------------------
// Small utility kernels
// ---------------------------------------------------------------------------
__global__ void zero_f32(float* p, int n) {
  int i = blockIdx.x * blockDim.x + threadIdx.x;
  if (i < n) p[i] = 0.0f;
}

// x16p[pack(n,k,32)] = x[n, k] (drop column 32 of 33)
__global__ void cvt_x(_Float16* __restrict__ d, const float* __restrict__ s) {
  int i = blockIdx.x * blockDim.x + threadIdx.x;
  if (i < NN * IN_DIM) {
    int n = i / IN_DIM, k = i % IN_DIM;
    d[a_pack_idx(n, k, IN_DIM)] = (_Float16)s[n * (IN_DIM + 1) + k];
  }
}

__global__ void deg_count(const int* __restrict__ ei, float* __restrict__ deg) {
  int e = blockIdx.x * blockDim.x + threadIdx.x;
  if (e < EE) atomicAdd(&deg[ei[EE + e]], 1.0f);
}

// m[dst] += mpre[src] per edge; 32 lanes cover D=64 (2 per lane)
__global__ void scatter_msg(const int* __restrict__ ei,
                            const float* __restrict__ mpre,
                            float* __restrict__ m) {
  int t = blockIdx.x * blockDim.x + threadIdx.x;
  int e = t >> 5;
  int l = t & 31;
  if (e < EE) {
    int src = ei[e];
    int dst = ei[EE + e];
    atomicAdd(&m[(size_t)dst * DD + l],      mpre[(size_t)src * DD + l]);
    atomicAdd(&m[(size_t)dst * DD + 32 + l], mpre[(size_t)src * DD + 32 + l]);
  }
}

// m16p = pack_f16(m / max(deg,1))
__global__ void scale_cvt(const float* __restrict__ m,
                          const float* __restrict__ deg,
                          _Float16* __restrict__ m16p) {
  int i = blockIdx.x * blockDim.x + threadIdx.x;
  if (i < NN * DD) {
    int n = i >> 6, d = i & 63;
    m16p[a_pack_idx(n, d, DD)] = (_Float16)(m[i] / fmaxf(deg[n], 1.0f));
  }
}

// GRU cell gates (biases folded in here); writes h (f32) + packed h16
__global__ void gru_gates(const float* __restrict__ gi, const float* __restrict__ gh,
                          const float* __restrict__ bih, const float* __restrict__ bhh,
                          float* __restrict__ h, _Float16* __restrict__ h16p) {
  int i = blockIdx.x * blockDim.x + threadIdx.x;
  if (i < NN * DD) {
    int n = i >> 6, d = i & 63;
    const float* gir = gi + (size_t)n * 3 * DD;
    const float* ghr = gh + (size_t)n * 3 * DD;
    float r  = sigm((gir[d]         + bih[d])        + (ghr[d]         + bhh[d]));
    float z  = sigm((gir[DD + d]    + bih[DD + d])   + (ghr[DD + d]    + bhh[DD + d]));
    float nn = tanhf((gir[2*DD + d] + bih[2*DD + d]) +
                     r * (ghr[2*DD + d] + bhh[2*DD + d]));
    float out = (1.0f - z) * nn + z * h[i];
    h[i] = out;
    h16p[a_pack_idx(n, d, DD)] = (_Float16)out;
  }
}

// Edge attention + aggregation, fused. One wave per dst row; ballot-scan of
// the adj row; wave-cooperative sigmoid-dot via butterfly shuffles.
__global__ void attention_agg(const float* __restrict__ af, const float* __restrict__ bf,
                              const float* __restrict__ ba1, const float* __restrict__ wa2,
                              const float* __restrict__ ba2, const int* __restrict__ adj,
                              const float* __restrict__ h, float* __restrict__ hnew) {
  const int dst  = blockIdx.x;
  const int lane = threadIdx.x;   // 32 lanes, 2 d-values each
  const float a0 = af[(size_t)dst * DD + lane];
  const float a1 = af[(size_t)dst * DD + 32 + lane];
  const float c0 = ba1[lane],  c1 = ba1[32 + lane];
  const float w0 = wa2[lane],  w1 = wa2[32 + lane];
  const float b2 = ba2[0];
  float acc0 = 0.0f, acc1 = 0.0f;
  const int* arow = adj + (size_t)dst * NN;
  for (int base = 0; base < NN; base += 32) {
    int av = arow[base + lane];
    unsigned long long mask = __ballot(av != 0);
    while (mask) {
      int j = __ffsll(mask) - 1;
      mask &= mask - 1;
      int src = base + j;
      const float* brow = bf + (size_t)src * DD;
      float s = sigm(a0 + brow[lane] + c0) * w0 +
                sigm(a1 + brow[32 + lane] + c1) * w1;
#pragma unroll
      for (int off = 16; off; off >>= 1) s += __shfl_xor(s, off, 32);
      float c = sigm(s + b2);
      const float* hrow = h + (size_t)src * DD;
      acc0 += c * hrow[lane];
      acc1 += c * hrow[32 + lane];
    }
  }
  hnew[(size_t)dst * DD + lane]      = acc0;
  hnew[(size_t)dst * DD + 32 + lane] = acc1;
}

__global__ void copy_cvt(const float* __restrict__ src, float* __restrict__ h,
                         _Float16* __restrict__ h16p) {
  int i = blockIdx.x * blockDim.x + threadIdx.x;
  if (i < NN * DD) {
    int n = i >> 6, d = i & 63;
    float v = src[i];
    h[i] = v;
    h16p[a_pack_idx(n, d, DD)] = (_Float16)v;
  }
}

// Two sigmoid heads: one wave per node, wave-reduced 64-dots
__global__ void heads(const float* __restrict__ t2,
                      const float* __restrict__ Wh1, const float* __restrict__ bh1,
                      const float* __restrict__ Wh2, const float* __restrict__ bh2,
                      float* __restrict__ out) {
  const int n = blockIdx.x;
  const int lane = threadIdx.x;
  float v0 = t2[(size_t)n * DD + lane];
  float v1 = t2[(size_t)n * DD + 32 + lane];
  float s1 = v0 * Wh1[lane] + v1 * Wh1[32 + lane];
  float s2 = v0 * Wh2[lane] + v1 * Wh2[32 + lane];
#pragma unroll
  for (int off = 16; off; off >>= 1) {
    s1 += __shfl_xor(s1, off, 32);
    s2 += __shfl_xor(s2, off, 32);
  }
  if (lane == 0) {
    out[n]      = sigm(s1 + bh1[0]);
    out[NN + n] = sigm(s2 + bh2[0]);
  }
}

// ---------------------------------------------------------------------------
// Host orchestration
// ---------------------------------------------------------------------------
static inline char* bump(char*& p, size_t bytes) {
  char* r = p;
  p += (bytes + 255) & ~(size_t)255;
  return r;
}

extern "C" void kernel_launch(void* const* d_in, const int* in_sizes, int n_in,
                              void* d_out, int out_size, void* d_ws, size_t ws_size,
                              hipStream_t stream) {
  (void)in_sizes; (void)n_in; (void)out_size; (void)ws_size;
  const float* x      = (const float*)d_in[0];
  const int*   ei     = (const int*)d_in[1];
  const int*   adj    = (const int*)d_in[2];
  const float* W_init = (const float*)d_in[3];
  const float* b_init = (const float*)d_in[4];
  const float* gru_w  = (const float*)d_in[5];
  const float* w_ih   = (const float*)d_in[6];
  const float* w_hh   = (const float*)d_in[7];
  const float* b_ih   = (const float*)d_in[8];
  const float* b_hh   = (const float*)d_in[9];
  const float* W_att1 = (const float*)d_in[10];
  const float* b_att1 = (const float*)d_in[11];
  const float* W_att2 = (const float*)d_in[12];
  const float* b_att2 = (const float*)d_in[13];
  const float* Wo_a   = (const float*)d_in[14];
  const float* bo_a   = (const float*)d_in[15];
  const float* Wo_b   = (const float*)d_in[16];
  const float* bo_b   = (const float*)d_in[17];
  const float* Wh1    = (const float*)d_in[18];
  const float* bh1    = (const float*)d_in[19];
  const float* Wh2    = (const float*)d_in[20];
  const float* bh2    = (const float*)d_in[21];
  float* out = (float*)d_out;

  // workspace carve-up
  char* p = (char*)d_ws;
  float* h     = (float*)bump(p, NN * DD * 4);
  float* hnew  = (float*)bump(p, NN * DD * 4);
  float* m     = (float*)bump(p, NN * DD * 4);
  float* mpre  = (float*)bump(p, NN * DD * 4);
  float* gi    = (float*)bump(p, NN * 3 * DD * 4);
  float* gh    = (float*)bump(p, NN * 3 * DD * 4);
  float* af    = (float*)bump(p, NN * DD * 4);
  float* bfb   = (float*)bump(p, NN * DD * 4);
  float* deg   = (float*)bump(p, NN * 4);
  _Float16* h16   = (_Float16*)bump(p, NN * DD * 2);       // A-fragment order
  _Float16* m16   = (_Float16*)bump(p, NN * DD * 2);       // A-fragment order
  _Float16* t16   = (_Float16*)bump(p, NN * DD * 2);       // A-fragment order
  _Float16* x16   = (_Float16*)bump(p, NN * IN_DIM * 2);   // A-fragment order
  _Float16* Wi16  = (_Float16*)bump(p, IN_DIM * DD * 2);   // B-fragment order
  _Float16* g016  = (_Float16*)bump(p, DD * DD * 2);
  _Float16* g116  = (_Float16*)bump(p, DD * DD * 2);
  _Float16* wihT  = (_Float16*)bump(p, DD * 3 * DD * 2);
  _Float16* whhT  = (_Float16*)bump(p, DD * 3 * DD * 2);
  _Float16* wa1lo = (_Float16*)bump(p, DD * DD * 2);
  _Float16* wa1hi = (_Float16*)bump(p, DD * DD * 2);
  _Float16* woa16 = (_Float16*)bump(p, DD * DD * 2);
  _Float16* wob16 = (_Float16*)bump(p, DD * DD * 2);

  auto g1d = [](int n) { return dim3((n + 255) / 256); };
  const dim3 b256(256);
  const dim3 wave(32);
  const dim3 gblk(128);                       // 4 WMMA waves / block
  auto gtiles = [](int M, int N) { return dim3(((M / 16) * (N / 16) + 3) / 4); };

  // --- weight / input packing into fragment order ---
  cvt_x<<<g1d(NN * IN_DIM), b256, 0, stream>>>(x16, x);
  pack_B<<<g1d(IN_DIM * DD), b256, 0, stream>>>(Wi16, W_init, IN_DIM, DD, 0);
  pack_B<<<g1d(DD * DD), b256, 0, stream>>>(g016, gru_w, DD, DD, 0);
  pack_B<<<g1d(DD * DD), b256, 0, stream>>>(g116, gru_w + DD * DD, DD, DD, 0);
  pack_B<<<g1d(DD * 3 * DD), b256, 0, stream>>>(wihT, w_ih, DD, 3 * DD, 1);
  pack_B<<<g1d(DD * 3 * DD), b256, 0, stream>>>(whhT, w_hh, DD, 3 * DD, 1);
  pack_B<<<g1d(DD * DD), b256, 0, stream>>>(wa1lo, W_att1, DD, DD, 0);
  pack_B<<<g1d(DD * DD), b256, 0, stream>>>(wa1hi, W_att1 + DD * DD, DD, DD, 0);
  pack_B<<<g1d(DD * DD), b256, 0, stream>>>(woa16, Wo_a, DD, DD, 0);
  pack_B<<<g1d(DD * DD), b256, 0, stream>>>(wob16, Wo_b, DD, DD, 0);

  // --- degree ---
  zero_f32<<<g1d(NN), b256, 0, stream>>>(deg, NN);
  deg_count<<<g1d(EE), b256, 0, stream>>>(ei, deg);

  // --- h = leaky(x @ W_init + b_init) : M=1024 N=64 K=32 ---
  gemm_wmma_f16<<<gtiles(NN, DD), gblk, 0, stream>>>(
      x16, Wi16, b_init, h, h16, NN, DD, IN_DIM, 1);

  for (int u = 0; u < 3; ++u) {
    for (int l = 0; l < 2; ++l) {
      // mpre = h @ gru_w[l]
      gemm_wmma_f16<<<gtiles(NN, DD), gblk, 0, stream>>>(
          h16, (l == 0) ? g016 : g116, nullptr, mpre, nullptr, NN, DD, DD, 0);
      // mean aggregation over edges
      zero_f32<<<g1d(NN * DD), b256, 0, stream>>>(m, NN * DD);
      scatter_msg<<<g1d(EE * 32), b256, 0, stream>>>(ei, mpre, m);
      scale_cvt<<<g1d(NN * DD), b256, 0, stream>>>(m, deg, m16);
      // gi = m @ w_ih^T ; gh = h @ w_hh^T  (N=192, K=64)
      gemm_wmma_f16<<<gtiles(NN, 3 * DD), gblk, 0, stream>>>(
          m16, wihT, nullptr, gi, nullptr, NN, 3 * DD, DD, 0);
      gemm_wmma_f16<<<gtiles(NN, 3 * DD), gblk, 0, stream>>>(
          h16, whhT, nullptr, gh, nullptr, NN, 3 * DD, DD, 0);
      gru_gates<<<g1d(NN * DD), b256, 0, stream>>>(gi, gh, b_ih, b_hh, h, h16);
    }
    // attention projections
    gemm_wmma_f16<<<gtiles(NN, DD), gblk, 0, stream>>>(
        h16, wa1lo, nullptr, af, nullptr, NN, DD, DD, 0);
    gemm_wmma_f16<<<gtiles(NN, DD), gblk, 0, stream>>>(
        h16, wa1hi, nullptr, bfb, nullptr, NN, DD, DD, 0);
    // fused sparse attention + aggregation (adj-masked, never N^2 x D)
    attention_agg<<<dim3(NN), wave, 0, stream>>>(af, bfb, b_att1, W_att2, b_att2,
                                                 adj, h, hnew);
    copy_cvt<<<g1d(NN * DD), b256, 0, stream>>>(hnew, h, h16);
  }

  // --- output MLP: t1 = leaky(h@Wo_a+bo_a); t2 = leaky(t1@Wo_b+bo_b) ---
  gemm_wmma_f16<<<gtiles(NN, DD), gblk, 0, stream>>>(
      h16, woa16, bo_a, mpre, t16, NN, DD, DD, 1);
  gemm_wmma_f16<<<gtiles(NN, DD), gblk, 0, stream>>>(
      t16, wob16, bo_b, m, nullptr, NN, DD, DD, 1);
  heads<<<dim3(NN), wave, 0, stream>>>(m, Wh1, bh1, Wh2, bh2, out);
}
